// MaskedGraphAttention_54408645706459
// MI455X (gfx1250) — matrix operand order repaired and checked by reading, not compiled, not run on hardware
//
#include <hip/hip_runtime.h>

// ---------------- problem constants ----------------
#define BATCH  4
#define SEQ    2048
#define DMODEL 512
#define NH     8
#define HD     64
#define D3     1536            // 3*DMODEL
#define MWORDS (SEQ / 32)      // 64 mask words per row
#define LOG2E  1.4426950408889634f
#define QSCALE (0.125f * LOG2E)   // 1/sqrt(HD) folded with log2(e) for exp2-domain softmax

#define DEV __device__ __forceinline__

typedef __bf16 bf16x16 __attribute__((ext_vector_type(16)));
typedef float  f32x8   __attribute__((ext_vector_type(8)));

union AFrag { bf16x16 v; uint4 q[2]; };

#if __has_builtin(__builtin_amdgcn_exp2f)
DEV float fast_exp2(float x) { return __builtin_amdgcn_exp2f(x); }   // v_exp_f32
#else
DEV float fast_exp2(float x) { return exp2f(x); }
#endif

// ---- WMMA bf16 16x16x32, f32 accumulate ----
DEV f32x8 wmma_bf16(bf16x16 a, bf16x16 b, f32x8 c) {
  return __builtin_amdgcn_wmma_f32_16x16x32_bf16(false, a, false, b, (short)0, c, false, false);
}

// A-fragment (16x32, MxK), row-major source, ld in elements.
// ISA layout: lanes 0-15 row M=lane, K chunks [0..7] in v0-3 and [16..23] in v4-7;
// lanes 16-31 row M=lane-16, K [8..15] and [24..31].
DEV bf16x16 load_a(const __bf16* base, int ld) {
  const int lane = threadIdx.x & 31;
  const int r = lane & 15, hf = lane >> 4;
  const __bf16* p = base + (size_t)r * ld + hf * 8;
  AFrag f;
  f.q[0] = *(const uint4*)(p);
  f.q[1] = *(const uint4*)(p + 16);
  return f.v;
}

// B-fragment (32x16, KxN) fed from an N x K (transposed) row-major source.
// ISA layout: lanes 0-15 col N=lane K [0..7] and [16..23]; lanes 16-31 col N=lane-16 K [8..15] and [24..31].
DEV bf16x16 load_b(const __bf16* baseT, int ld) {
  const int lane = threadIdx.x & 31;
  const int r = lane & 15, hf = lane >> 4;
  const __bf16* p = baseT + (size_t)r * ld + hf * 8;
  AFrag f;
  f.q[0] = *(const uint4*)(p);
  f.q[1] = *(const uint4*)(p + 16);
  return f.v;
}

// ---------------- kernel 0: fp32 -> bf16 convert ----------------
__global__ void cvt_bf16_kernel(const float* __restrict__ in, __bf16* __restrict__ out, int n) {
  int i = blockIdx.x * 256 + threadIdx.x;
  if (i < n) out[i] = (__bf16)in[i];
}

// ---------------- kernel 0b: transpose (K x Ncols) fp32 -> (Ncols x K) bf16 ----------------
__global__ void transpose_w_kernel(const float* __restrict__ in, __bf16* __restrict__ out,
                                   int rows /*K*/, int cols /*N*/) {
  int i = blockIdx.x * 256 + threadIdx.x;
  if (i < rows * cols) {
    int c = i / rows, r = i % rows;          // out index i = c*rows + r (coalesced writes)
    out[i] = (__bf16)in[(size_t)r * cols + c];
  }
}

// ---------------- kernel 0c: bit-pack bool mask -> u32 words (32 keys/word) ----------------
__global__ void mask_pack_kernel(const unsigned char* __restrict__ in,
                                 unsigned int* __restrict__ out, int nwords) {
  int i = blockIdx.x * 256 + threadIdx.x;
  if (i >= nwords) return;
  const uint4* q = (const uint4*)(in + (size_t)i * 32);
  uint4 a = q[0], c = q[1];
  unsigned int src[8] = {a.x, a.y, a.z, a.w, c.x, c.y, c.z, c.w};
  unsigned int w = 0;
  #pragma unroll
  for (int j = 0; j < 8; ++j) {
    unsigned int v = src[j];
    w |= ((v & 0x000000FFu) ? 1u : 0u) << (j * 4 + 0);
    w |= ((v & 0x0000FF00u) ? 1u : 0u) << (j * 4 + 1);
    w |= ((v & 0x00FF0000u) ? 1u : 0u) << (j * 4 + 2);
    w |= ((v & 0xFF000000u) ? 1u : 0u) << (j * 4 + 3);
  }
  out[i] = w;
}

// ---------------- kernel 1: QKV GEMM (8192 x 1536) ----------------
__global__ __launch_bounds__(256)
void qkv_gemm_kernel(const __bf16* __restrict__ Xb, const __bf16* __restrict__ WT,
                     __bf16* __restrict__ Qh, __bf16* __restrict__ Kh,
                     __bf16* __restrict__ Vt) {
  const int lane = threadIdx.x & 31;
  const int wave = threadIdx.x >> 5;                 // 0..7
  const int wr = wave & 3, wc = wave >> 2;           // 4 x 2 wave grid
  const int m0 = blockIdx.y * 128 + wr * 32;
  const int n0 = blockIdx.x * 64 + wc * 32;

  f32x8 acc[2][2] = {};
  for (int kk = 0; kk < DMODEL; kk += 32) {
    bf16x16 a0 = load_a(Xb + (size_t)m0 * DMODEL + kk, DMODEL);
    bf16x16 a1 = load_a(Xb + (size_t)(m0 + 16) * DMODEL + kk, DMODEL);
    bf16x16 b0 = load_b(WT + (size_t)n0 * DMODEL + kk, DMODEL);
    bf16x16 b1 = load_b(WT + (size_t)(n0 + 16) * DMODEL + kk, DMODEL);
    acc[0][0] = wmma_bf16(a0, b0, acc[0][0]);
    acc[0][1] = wmma_bf16(a0, b1, acc[0][1]);
    acc[1][0] = wmma_bf16(a1, b0, acc[1][0]);
    acc[1][1] = wmma_bf16(a1, b1, acc[1][1]);
  }

  const int mb = (lane >> 4) * 8, nn = lane & 15;    // C layout: 8 rows per lane
  #pragma unroll
  for (int mi = 0; mi < 2; ++mi)
    #pragma unroll
    for (int ni = 0; ni < 2; ++ni)
      #pragma unroll
      for (int i = 0; i < 8; ++i) {
        int gm = m0 + mi * 16 + mb + i;              // token index 0..8191
        int gc = n0 + ni * 16 + nn;                  // qkv column 0..1535
        float v = acc[mi][ni][i];
        int b = gm >> 11, n = gm & (SEQ - 1);
        int sec = gc >> 9, c = gc & (DMODEL - 1);
        int h = c >> 6, dk = c & (HD - 1);
        size_t bh = (size_t)(b * NH + h);
        if (sec == 0)      Qh[(bh * SEQ + n) * HD + dk] = (__bf16)(v * QSCALE);
        else if (sec == 1) Kh[(bh * SEQ + n) * HD + dk] = (__bf16)v;
        else               Vt[(bh * HD + dk) * SEQ + n] = (__bf16)v;
      }
}

// ---------------- kernel 2: flash attention ----------------
// grid (SEQ/64, BATCH*NH), 128 threads (4 waves x 16 query rows).
__global__ __launch_bounds__(128)
void attn_kernel(const __bf16* __restrict__ Qh, const __bf16* __restrict__ Kh,
                 const __bf16* __restrict__ Vt, const unsigned int* __restrict__ Mbits,
                 __bf16* __restrict__ AO) {
  __shared__ __align__(16) __bf16 pbuf[4][16 * 32];  // per-wave P tile (16x32)

  const int lane = threadIdx.x & 31;
  const int wave = threadIdx.x >> 5;                 // 0..3
  const int bh = blockIdx.y;                         // 0..31
  const int b = bh >> 3, h = bh & (NH - 1);
  const int q0 = blockIdx.x * 64 + wave * 16;
  const int r = lane & 15, mb = (lane >> 4) * 8;

  const __bf16* Qbase = Qh + (size_t)bh * SEQ * HD;
  const __bf16* Kbase = Kh + (size_t)bh * SEQ * HD;
  const __bf16* Vbase = Vt + (size_t)bh * HD * SEQ;
  const unsigned int* mwrow = Mbits + ((size_t)b * SEQ + q0) * MWORDS;

  // Q A-fragments for the whole row block (contraction dk = 2 x 32)
  bf16x16 aq0 = load_a(Qbase + (size_t)q0 * HD + 0,  HD);
  bf16x16 aq1 = load_a(Qbase + (size_t)q0 * HD + 32, HD);

  f32x8 o[4] = {};
  float rm[8], rl[8];
  #pragma unroll
  for (int i = 0; i < 8; ++i) { rm[i] = -1e30f; rl[i] = 0.f; }

  __bf16* pb = &pbuf[wave][0];

  for (int k0 = 0; k0 < SEQ; k0 += 32) {
    // ---- prefetch next key block (K: 32 rows x 128B; V: 64 rows x 64B) ----
    if (k0 + 32 < SEQ) {
      __builtin_prefetch((const void*)(Kbase + (size_t)(k0 + 32 + lane) * HD), 0, 3);
      __builtin_prefetch((const void*)(Vbase + (size_t)lane * SEQ + (k0 + 32)), 0, 3);
      __builtin_prefetch((const void*)(Vbase + (size_t)(32 + lane) * SEQ + (k0 + 32)), 0, 3);
    }
    // ---- S = Q * K^T for 32 keys (two 16x16 C tiles) ----
    f32x8 s0 = {}, s1 = {};
    {
      bf16x16 bk;
      bk = load_b(Kbase + (size_t)k0 * HD + 0,  HD);        s0 = wmma_bf16(aq0, bk, s0);
      bk = load_b(Kbase + (size_t)k0 * HD + 32, HD);        s0 = wmma_bf16(aq1, bk, s0);
      bk = load_b(Kbase + (size_t)(k0 + 16) * HD + 0,  HD); s1 = wmma_bf16(aq0, bk, s1);
      bk = load_b(Kbase + (size_t)(k0 + 16) * HD + 32, HD); s1 = wmma_bf16(aq1, bk, s1);
    }
    // ---- mask from packed bits (1 = keep); one u32 covers this row's 32 keys ----
    const int kw = k0 >> 5;
    #pragma unroll
    for (int i = 0; i < 8; ++i) {
      unsigned int wv = mwrow[(size_t)(mb + i) * MWORDS + kw];
      if (!((wv >> r) & 1u))        s0[i] = -1e30f;
      if (!((wv >> (16 + r)) & 1u)) s1[i] = -1e30f;
    }
    // ---- online softmax (exp2 domain; Q pre-scaled by 1/sqrt(dk)*log2e) ----
    #pragma unroll
    for (int i = 0; i < 8; ++i) {
      float cm = fmaxf(s0[i], s1[i]);
      cm = fmaxf(cm, __shfl_xor(cm, 1, 32));
      cm = fmaxf(cm, __shfl_xor(cm, 2, 32));
      cm = fmaxf(cm, __shfl_xor(cm, 4, 32));
      cm = fmaxf(cm, __shfl_xor(cm, 8, 32));
      float mn = fmaxf(rm[i], cm);
      float alpha = fast_exp2(rm[i] - mn);
      rm[i] = mn;
      float p0 = fast_exp2(s0[i] - mn);
      float p1 = fast_exp2(s1[i] - mn);
      float cs = p0 + p1;
      cs += __shfl_xor(cs, 1, 32);
      cs += __shfl_xor(cs, 2, 32);
      cs += __shfl_xor(cs, 4, 32);
      cs += __shfl_xor(cs, 8, 32);
      rl[i] = rl[i] * alpha + cs;
      #pragma unroll
      for (int t = 0; t < 4; ++t) o[t][i] *= alpha;
      // C-layout -> row-major LDS (the A-layout transpose bounce)
      pb[(mb + i) * 32 + r]      = (__bf16)p0;
      pb[(mb + i) * 32 + 16 + r] = (__bf16)p1;
    }
    asm volatile("s_wait_dscnt 0" ::: "memory");   // wave-private LDS: stores visible before loads
    bf16x16 ap = load_a(pb, 32);                   // P as 16x32 A-fragment
    // ---- O += P * V (V stored transposed: contiguous B-fragments) ----
    #pragma unroll
    for (int t = 0; t < 4; ++t) {
      bf16x16 bv = load_b(Vbase + (size_t)(t * 16) * SEQ + k0, SEQ);
      o[t] = wmma_bf16(ap, bv, o[t]);
    }
  }

  // ---- normalize and write (B,N,D) bf16 ----
  #pragma unroll
  for (int t = 0; t < 4; ++t)
    #pragma unroll
    for (int i = 0; i < 8; ++i) {
      int row = q0 + mb + i;
      int col = h * HD + t * 16 + r;
      AO[((size_t)b * SEQ + row) * DMODEL + col] = (__bf16)(o[t][i] / rl[i]);
    }
}

// ---------------- kernel 3: output projection + bias + LeakyReLU ----------------
__global__ __launch_bounds__(256)
void out_gemm_kernel(const __bf16* __restrict__ AO, const __bf16* __restrict__ WT,
                     const float* __restrict__ bias, float* __restrict__ out) {
  const int lane = threadIdx.x & 31;
  const int wave = threadIdx.x >> 5;
  const int wr = wave & 3, wc = wave >> 2;
  const int m0 = blockIdx.y * 128 + wr * 32;
  const int n0 = blockIdx.x * 64 + wc * 32;

  f32x8 acc[2][2] = {};
  for (int kk = 0; kk < DMODEL; kk += 32) {
    bf16x16 a0 = load_a(AO + (size_t)m0 * DMODEL + kk, DMODEL);
    bf16x16 a1 = load_a(AO + (size_t)(m0 + 16) * DMODEL + kk, DMODEL);
    bf16x16 b0 = load_b(WT + (size_t)n0 * DMODEL + kk, DMODEL);
    bf16x16 b1 = load_b(WT + (size_t)(n0 + 16) * DMODEL + kk, DMODEL);
    acc[0][0] = wmma_bf16(a0, b0, acc[0][0]);
    acc[0][1] = wmma_bf16(a0, b1, acc[0][1]);
    acc[1][0] = wmma_bf16(a1, b0, acc[1][0]);
    acc[1][1] = wmma_bf16(a1, b1, acc[1][1]);
  }

  const int mb = (lane >> 4) * 8, nn = lane & 15;
  #pragma unroll
  for (int mi = 0; mi < 2; ++mi)
    #pragma unroll
    for (int ni = 0; ni < 2; ++ni)
      #pragma unroll
      for (int i = 0; i < 8; ++i) {
        int gm = m0 + mi * 16 + mb + i;
        int gc = n0 + ni * 16 + nn;
        float v = acc[mi][ni][i] + bias[gc];
        out[(size_t)gm * DMODEL + gc] = (v >= 0.f) ? v : 0.1f * v;
      }
}

// ---------------- workspace layout (bytes) ----------------
static constexpr size_t XB_OFF    = 0;                          // 8192*512 bf16  = 8 MB
static constexpr size_t WQKVT_OFF = XB_OFF    + 8388608;        // 1536*512 bf16  = 1.5 MB
static constexpr size_t WOUTT_OFF = WQKVT_OFF + 1572864;        // 512*512 bf16   = 0.5 MB
static constexpr size_t QH_OFF    = WOUTT_OFF + 524288;         // 8 MB
static constexpr size_t KH_OFF    = QH_OFF    + 8388608;        // 8 MB
static constexpr size_t VT_OFF    = KH_OFF    + 8388608;        // 8 MB
static constexpr size_t AO_OFF    = VT_OFF    + 8388608;        // 8 MB
static constexpr size_t MB_OFF    = AO_OFF    + 8388608;        // packed mask = 2 MB (total ~44 MB)

extern "C" void kernel_launch(void* const* d_in, const int* in_sizes, int n_in,
                              void* d_out, int out_size, void* d_ws, size_t ws_size,
                              hipStream_t stream) {
  const float* x            = (const float*)d_in[0];
  const unsigned char* mask = (const unsigned char*)d_in[1];   // jnp bool = 1 byte/elem
  const float* Wqkv         = (const float*)d_in[2];
  const float* Wout         = (const float*)d_in[3];
  const float* bout         = (const float*)d_in[4];
  float* out                = (float*)d_out;

  char* ws = (char*)d_ws;
  __bf16* Xb          = (__bf16*)(ws + XB_OFF);
  __bf16* WqkvT       = (__bf16*)(ws + WQKVT_OFF);
  __bf16* WoutT       = (__bf16*)(ws + WOUTT_OFF);
  __bf16* Qh          = (__bf16*)(ws + QH_OFF);
  __bf16* Kh          = (__bf16*)(ws + KH_OFF);
  __bf16* Vt          = (__bf16*)(ws + VT_OFF);
  __bf16* AO          = (__bf16*)(ws + AO_OFF);
  unsigned int* Mbits = (unsigned int*)(ws + MB_OFF);

  const int nx = BATCH * SEQ * DMODEL;                 // 4,194,304
  const int nw = BATCH * SEQ * MWORDS;                 // 524,288 mask words
  cvt_bf16_kernel<<<(nx + 255) / 256, 256, 0, stream>>>(x, Xb, nx);
  transpose_w_kernel<<<(DMODEL * D3 + 255) / 256, 256, 0, stream>>>(Wqkv, WqkvT, DMODEL, D3);
  transpose_w_kernel<<<(DMODEL * DMODEL + 255) / 256, 256, 0, stream>>>(Wout, WoutT, DMODEL, DMODEL);
  mask_pack_kernel<<<(nw + 255) / 256, 256, 0, stream>>>(mask, Mbits, nw);

  qkv_gemm_kernel<<<dim3(D3 / 64, (BATCH * SEQ) / 128), 256, 0, stream>>>(Xb, WqkvT, Qh, Kh, Vt);
  attn_kernel<<<dim3(SEQ / 64, BATCH * NH), 128, 0, stream>>>(Qh, Kh, Vt, Mbits, AO);
  out_gemm_kernel<<<dim3(DMODEL / 64, (BATCH * SEQ) / 128), 256, 0, stream>>>(AO, WoutT, bout, out);
}
